// DeepFCNet_4526895530380
// MI455X (gfx1250) — matrix operand order repaired and checked by reading, not compiled, not run on hardware
//
#include <hip/hip_runtime.h>
#include <hip/hip_bf16.h>
#include <math.h>

// ---------------------------------------------------------------------------
// DeepFCNet forward for MI455X (gfx1250, wave32, WMMA).
// All GEMM-shaped layers run on v_wmma_f32_16x16x32_f16 (f16 in, f32 accum).
// Conv activation tiles are staged into LDS with global_load_async_to_lds_b128
// (ASYNCcnt path) instead of the VGPR round-trip.
// ---------------------------------------------------------------------------

typedef _Float16 half_t;
typedef __attribute__((ext_vector_type(16))) _Float16 v16h;
typedef __attribute__((ext_vector_type(8)))  _Float16 v8h;
typedef __attribute__((ext_vector_type(8)))  float    v8f;

#define NETS_C   135
#define NNET     8640           // 64 * 135
#define NPAIRS   9045           // 135*134/2
#define BATCH    64
#define BN_INV   0.9999950000374997f   // 1/sqrt(1 + 1e-5)
#define RR_SLOPE 0.22916666666666666f  // eval-mode RReLU negative slope

// ---------------- WMMA fragment helpers (contiguous k order, same in A and B) -----

__device__ __forceinline__ v8f wmma_f16(v16h a, v16h b, v8f c) {
  return __builtin_amdgcn_wmma_f32_16x16x32_f16(
      /*neg_a=*/false, a, /*neg_b=*/false, b,
      /*c_mod=*/(short)0, c, /*reuse_a=*/false, /*reuse_b=*/false);
}

// A 16x32 f16: lane holds row M=lane%16; element e -> K = e + 16*(lane>=16)
__device__ __forceinline__ v16h load_frag_a(const half_t* rowPtr, int kbase, int laneHi) {
  return *(const v16h*)(rowPtr + kbase + (laneHi ? 16 : 0));
}

// B 32x16 f16: lane holds col N=lane%16; element e -> K = e + 16*(lane>=16)
__device__ __forceinline__ v16h load_frag_b(const half_t* rowPtr, int kbase, int laneHi) {
  return *(const v16h*)(rowPtr + kbase + (laneHi ? 16 : 0));
}

// Build a B fragment from an im2col view of an LDS activation tile.
// K index decomposes as (cin, kk): value = sh[cin*width + pos + kk].
__device__ __forceinline__ v16h im2col_b(const half_t* sh, int width, int Ktot,
                                         int pos, int kbase, int laneHi) {
  v16h b;
#pragma unroll
  for (int e = 0; e < 16; ++e) {
    int K = kbase + laneHi * 16 + e;
    _Float16 v = (_Float16)0.f;
    if (K < Ktot) {
      int cin = K / 3;
      int kk  = K - cin * 3;
      int w   = pos + kk;
      if (w < width) v = sh[cin * width + w];
    }
    b[e] = v;
  }
  return b;
}

// ---------------- async global -> LDS copy (CDNA5 ASYNCcnt path) -----------
// 16B per lane per issue; EXEC-masked per lane. LDS operand is the low 32 bits
// of the flat shared pointer (aperture lives in the high half per ISA 10.2).

__device__ __forceinline__ void async_copy_16B(const half_t* lds_dst,
                                               const half_t* gsrc) {
  unsigned int lds_off = (unsigned int)(unsigned long long)lds_dst;
  asm volatile("global_load_async_to_lds_b128 %0, %1, off"
               :: "v"(lds_off), "v"(gsrc)
               : "memory");
}

__device__ __forceinline__ void async_wait0() {
  asm volatile("s_wait_asynccnt 0" ::: "memory");
}

// ---------------- prep kernels ---------------------------------------------

// fp32 [R][C] -> f16 [R][Cp] with zero pad in K so WMMA k-loops are uniform
__global__ __launch_bounds__(256) void cvt_pad_kernel(const float* __restrict__ src,
                                                      half_t* __restrict__ dst,
                                                      int R, int C, int Cp) {
  int idx = blockIdx.x * 256 + threadIdx.x;
  if (idx >= R * Cp) return;
  int r = idx / Cp, c = idx - r * Cp;
  dst[idx] = (c < C) ? (half_t)src[(size_t)r * C + c] : (half_t)0.f;
}

// triu(k=1) pair index table: p -> (i, j), row-major as in np.triu_indices
__global__ __launch_bounds__(256) void pair_idx_kernel(int* __restrict__ ii,
                                                       int* __restrict__ jj) {
  int p = blockIdx.x * 256 + threadIdx.x;
  if (p >= NPAIRS) return;
  int i = 0, rem = p, cnt = NETS_C - 1;
  while (rem >= cnt) { rem -= cnt; ++i; --cnt; }
  ii[p] = i;
  jj[p] = i + 1 + rem;
}

// zero the 2160..2175 pad tail of each 2176-wide h3 row
__global__ __launch_bounds__(256) void h3_pad_kernel(half_t* __restrict__ h3) {
  int t = blockIdx.x * 256 + threadIdx.x;
  if (t >= NNET * 16) return;
  h3[(size_t)(t >> 4) * 2176 + 2160 + (t & 15)] = (half_t)0.f;
}

// zero the 9045..9055 pad tail of each 9056-wide s row
__global__ __launch_bounds__(64) void s_pad_kernel(half_t* __restrict__ s) {
  int t = blockIdx.x * 64 + threadIdx.x;
  if (t >= BATCH * 11) return;
  s[(size_t)(t / 11) * 9056 + 9045 + (t % 11)] = (half_t)0.f;
}

// ---------------- stage 1: conv1 (Cin=1,K=3) + bn + leaky + pool2 ----------
// 3 MACs/output -> VALU. Output h1: [net][16][186] f16.
__global__ __launch_bounds__(256)
void conv1_kernel(const float* __restrict__ x, const float* __restrict__ w,
                  const float* __restrict__ cb, const float* __restrict__ g,
                  const float* __restrict__ bb, half_t* __restrict__ h1) {
  int t = blockIdx.x * 256 + threadIdx.x;
  if (t >= NNET * 16 * 186) return;
  int net = t / (16 * 186);
  int rem = t - net * (16 * 186);
  int c2  = rem / 186;
  int w2  = rem - c2 * 186;
  const float* xr = x + (size_t)net * 375 + 2 * w2;
  float x0 = xr[0], x1 = xr[1], x2 = xr[2], x3 = xr[3];
  float best = -1e30f;
#pragma unroll
  for (int dc = 0; dc < 2; ++dc) {
    int co = 2 * c2 + dc;
    float k0 = w[co * 3], k1 = w[co * 3 + 1], k2 = w[co * 3 + 2];
    float sc = g[co] * BN_INV, sh = bb[co], bias = cb[co];
    float y0 = sc * (x0 * k0 + x1 * k1 + x2 * k2 + bias) + sh;
    float y1 = sc * (x1 * k0 + x2 * k1 + x3 * k2 + bias) + sh;
    y0 = (y0 >= 0.f) ? y0 : 0.01f * y0;
    y1 = (y1 >= 0.f) ? y1 : 0.01f * y1;
    best = fmaxf(best, fmaxf(y0, y1));
  }
  h1[(size_t)net * 2976 + c2 * 186 + w2] = (half_t)best;
}

// ---------------- stages 2/3: conv-as-GEMM (WMMA) + bn + leaky + pool2 -----
// A = weights [COUT x CIN*3] f16 (K padded to KCH*32), B = im2col from LDS.
// One block per net; wave w owns cout tile [16w, 16w+16). Activation tile is
// filled with async global->LDS b128 copies (overlaps the weight A loads).
template <int CIN, int WIN, int COUT, int NT, int KCH, int POOLW>
__global__ __launch_bounds__(32 * (COUT / 16))
void conv_bn_pool_kernel(const half_t* __restrict__ hin, const half_t* __restrict__ wh,
                         const float* __restrict__ cb, const float* __restrict__ g,
                         const float* __restrict__ bb, half_t* __restrict__ hout,
                         int outStride) {
  constexpr int WAVES  = COUT / 16;
  constexpr int KTOT   = CIN * 3;
  constexpr int KP     = KCH * 32;
  constexpr int TILE16 = (CIN * WIN) / 8;   // tile size in 16B chunks
  static_assert((CIN * WIN) % 8 == 0, "tile must be a multiple of 16B");
  __shared__ half_t sh[CIN * WIN];
  __shared__ float  stg[WAVES][16][16];

  const int net = blockIdx.x;
  const int tid = threadIdx.x;
  // async fill of the LDS activation tile (ASYNCcnt path)
  for (int i = tid; i < TILE16; i += 32 * WAVES)
    async_copy_16B(sh + i * 8, hin + (size_t)net * (CIN * WIN) + i * 8);

  const int wave = tid >> 5, lane = tid & 31, m = lane & 15, laneHi = lane >> 4;
  const int coutA = wave * 16 + m;
  v16h a[KCH];
#pragma unroll
  for (int kc = 0; kc < KCH; ++kc)
    a[kc] = load_frag_a(wh + (size_t)coutA * KP, kc * 32, laneHi);

  async_wait0();
  __syncthreads();

  for (int nt = 0; nt < NT; ++nt) {
    const int pos = nt * 16 + m;  // B column = conv output position
    v8f acc = {};
#pragma unroll
    for (int kc = 0; kc < KCH; ++kc) {
      v16h b = im2col_b(sh, WIN, KTOT, pos, kc * 32, laneHi);
      acc = wmma_f16(a[kc], b, acc);
    }
    // fused conv-bias + BN + leaky BEFORE pooling (bn gain may be negative)
#pragma unroll
    for (int r = 0; r < 8; ++r) {
      int co = wave * 16 + r + laneHi * 8;
      float v = g[co] * BN_INV * (acc[r] + cb[co]) + bb[co];
      v = (v >= 0.f) ? v : 0.01f * v;
      stg[wave][r + laneHi * 8][m] = v;
    }
    __syncthreads();
    // MaxPool2d(2) over (C, W) within this 16x16 tile -> 8x8 pooled outputs
    for (int t = lane; t < 64; t += 32) {
      int pc = t >> 3, pw = t & 7;
      float v = fmaxf(fmaxf(stg[wave][2 * pc][2 * pw],     stg[wave][2 * pc][2 * pw + 1]),
                      fmaxf(stg[wave][2 * pc + 1][2 * pw], stg[wave][2 * pc + 1][2 * pw + 1]));
      int c2 = wave * 8 + pc, w2 = nt * 8 + pw;
      if (w2 < POOLW)
        hout[(size_t)net * outStride + c2 * POOLW + w2] = (half_t)v;
    }
    __syncthreads();
  }
}

// ---------------- stage 4: fe linear (2160 -> 32), WMMA --------------------
// A = h3 rows (nets), B = fe_w rows (outputs). 540 blocks x 2 waves.
__global__ __launch_bounds__(64)
void fe_kernel(const half_t* __restrict__ h3, const half_t* __restrict__ fw,
               const float* __restrict__ fb, half_t* __restrict__ feats) {
  const int tid = threadIdx.x;
  const int wave = tid >> 5, lane = tid & 31, m = lane & 15, laneHi = lane >> 4;
  const half_t* arow = h3 + (size_t)(blockIdx.x * 16 + m) * 2176;
  const half_t* brow = fw + (size_t)(wave * 16 + m) * 2176;
  v8f acc = {};
  for (int kc = 0; kc < 68; ++kc) {
    v16h a = load_frag_a(arow, kc * 32, laneHi);
    v16h b = load_frag_b(brow, kc * 32, laneHi);
    acc = wmma_f16(a, b, acc);
  }
#pragma unroll
  for (int r = 0; r < 8; ++r) {
    int net = blockIdx.x * 16 + r + laneHi * 8;
    int n   = wave * 16 + m;
    feats[(size_t)net * 32 + n] = (half_t)(acc[r] + fb[n]);
  }
}

// ---------------- stage 5: pair MLP ----------------------------------------
// s1 (concat 64 -> 32) on WMMA; s2/s3/s4 scalar from LDS; tanh; f16 out.
__global__ __launch_bounds__(64)
void pair_mlp_kernel(const half_t* __restrict__ feats,
                     const int* __restrict__ ii, const int* __restrict__ jj,
                     const half_t* __restrict__ s1w, const float* __restrict__ s1b,
                     const float* __restrict__ s2w, const float* __restrict__ s2b,
                     const float* __restrict__ s3w, const float* __restrict__ s3b,
                     const float* __restrict__ s4w, const float* __restrict__ s4b,
                     half_t* __restrict__ s_out) {
  __shared__ float s1o[16][32];
  const int b = blockIdx.y;
  const int tid = threadIdx.x;
  const int wave = tid >> 5, lane = tid & 31, m = lane & 15, laneHi = lane >> 4;

  int p  = blockIdx.x * 16 + m;
  int pv = (p < NPAIRS) ? p : (NPAIRS - 1);
  const half_t* fi = feats + ((size_t)b * NETS_C + ii[pv]) * 32;
  const half_t* fj = feats + ((size_t)b * NETS_C + jj[pv]) * 32;
  // A row = concat(feats[i], feats[j]); k-chunk 0 = i half, chunk 1 = j half
  v16h a0 = load_frag_a(fi, 0, laneHi);
  v16h a1 = load_frag_a(fj, 0, laneHi);
  const half_t* brow = s1w + (size_t)(wave * 16 + m) * 64;
  v16h b0 = load_frag_b(brow, 0, laneHi);
  v16h b1 = load_frag_b(brow, 32, laneHi);
  v8f acc = {};
  acc = wmma_f16(a0, b0, acc);
  acc = wmma_f16(a1, b1, acc);
#pragma unroll
  for (int r = 0; r < 8; ++r) {
    float v = acc[r] + s1b[wave * 16 + m];
    s1o[r + laneHi * 8][wave * 16 + m] = (v > 0.f) ? v : 0.f;
  }
  __syncthreads();
  if (tid < 16) {
    int p2 = blockIdx.x * 16 + tid;
    if (p2 < NPAIRS) {
      float v2[16];
#pragma unroll
      for (int o = 0; o < 16; ++o) {
        float a = s2b[o];
        for (int k = 0; k < 32; ++k) a += s1o[tid][k] * s2w[o * 32 + k];
        v2[o] = (a > 0.f) ? a : 0.f;
      }
      float v3[8];
#pragma unroll
      for (int o = 0; o < 8; ++o) {
        float a = s3b[o];
        for (int k = 0; k < 16; ++k) a += v2[k] * s3w[o * 16 + k];
        v3[o] = (a > 0.f) ? a : 0.f;
      }
      float a4 = s4b[0];
#pragma unroll
      for (int k = 0; k < 8; ++k) a4 += v3[k] * s4w[k];
      s_out[(size_t)b * 9056 + p2] = (half_t)tanhf(a4);
    }
  }
}

// ---------------- stage 6: classification GEMMs (c1/c2/c3), WMMA ----------
// M = 64 (4 tiles, blockIdx.y); 4 waves/block each own one N tile.
// slope: RR_SLOPE for c1 (eval RReLU), 0 for relu layers.
__global__ __launch_bounds__(128)
void mlp_gemm_kernel(const half_t* __restrict__ A, const half_t* __restrict__ B,
                     const float* __restrict__ bias, half_t* __restrict__ out,
                     int Kp, int kch, int N, float slope) {
  const int tid = threadIdx.x;
  const int wave = tid >> 5, lane = tid & 31, m = lane & 15, laneHi = lane >> 4;
  const int ntile = blockIdx.x * 4 + wave;
  const int mtile = blockIdx.y;
  const half_t* arow = A + (size_t)(mtile * 16 + m) * Kp;
  const half_t* brow = B + (size_t)(ntile * 16 + m) * Kp;
  v8f acc = {};
  for (int kc = 0; kc < kch; ++kc) {
    int kb = kc * 32;
    if (kb + 160 <= Kp) {  // stream-ahead hints -> global_prefetch_b8
      __builtin_prefetch(arow + kb + 128, 0, 0);
      __builtin_prefetch(brow + kb + 128, 0, 0);
    }
    v16h a = load_frag_a(arow, kb, laneHi);
    v16h b = load_frag_b(brow, kb, laneHi);
    acc = wmma_f16(a, b, acc);
  }
#pragma unroll
  for (int r = 0; r < 8; ++r) {
    int row = mtile * 16 + r + laneHi * 8;
    int n   = ntile * 16 + m;
    float v = acc[r] + bias[n];
    v = (v >= 0.f) ? v : slope * v;
    out[(size_t)row * N + n] = (half_t)v;
  }
}

// ---------------- stage 7: c4 (64 -> 3) + log_softmax ----------------------
__global__ __launch_bounds__(64)
void head_kernel(const half_t* __restrict__ c3o, const float* __restrict__ w,
                 const float* __restrict__ bsc, float* __restrict__ out) {
  int row = threadIdx.x;
  float l[3];
#pragma unroll
  for (int j = 0; j < 3; ++j) {
    float a = bsc[j];
    for (int k = 0; k < 64; ++k) a += (float)c3o[row * 64 + k] * w[j * 64 + k];
    l[j] = a;
  }
  float mx  = fmaxf(l[0], fmaxf(l[1], l[2]));
  float lse = logf(expf(l[0] - mx) + expf(l[1] - mx) + expf(l[2] - mx));
#pragma unroll
  for (int j = 0; j < 3; ++j) out[row * 3 + j] = l[j] - mx - lse;
}

// ---------------------------------------------------------------------------

extern "C" void kernel_launch(void* const* d_in, const int* in_sizes, int n_in,
                              void* d_out, int out_size, void* d_ws, size_t ws_size,
                              hipStream_t stream) {
  (void)in_sizes; (void)n_in; (void)out_size; (void)ws_size;
  const float* x   = (const float*)d_in[0];
  const float* cw1 = (const float*)d_in[1];  const float* cb1 = (const float*)d_in[2];
  const float* g1  = (const float*)d_in[3];  const float* bb1 = (const float*)d_in[4];
  const float* cw2 = (const float*)d_in[5];  const float* cb2 = (const float*)d_in[6];
  const float* g2  = (const float*)d_in[7];  const float* bb2 = (const float*)d_in[8];
  const float* cw3 = (const float*)d_in[9];  const float* cb3 = (const float*)d_in[10];
  const float* g3  = (const float*)d_in[11]; const float* bb3 = (const float*)d_in[12];
  const float* few = (const float*)d_in[13]; const float* feb = (const float*)d_in[14];
  const float* s1w = (const float*)d_in[15]; const float* s1b = (const float*)d_in[16];
  const float* s2w = (const float*)d_in[17]; const float* s2b = (const float*)d_in[18];
  const float* s3w = (const float*)d_in[19]; const float* s3b = (const float*)d_in[20];
  const float* s4w = (const float*)d_in[21]; const float* s4b = (const float*)d_in[22];
  const float* C1w = (const float*)d_in[23]; const float* C1b = (const float*)d_in[24];
  const float* C2w = (const float*)d_in[25]; const float* C2b = (const float*)d_in[26];
  const float* C3w = (const float*)d_in[27]; const float* C3b = (const float*)d_in[28];
  const float* C4w = (const float*)d_in[29]; const float* C4b = (const float*)d_in[30];
  float* out = (float*)d_out;

  // ---- workspace bump allocator (256B aligned) ----
  char* base = (char*)d_ws;
  size_t off = 0;
  auto alloc = [&](size_t bytes) -> void* {
    void* p = base + off;
    off += (bytes + 255) & ~(size_t)255;
    return p;
  };
  half_t* h1    = (half_t*)alloc((size_t)NNET * 2976 * 2);  // [net][16][186]
  half_t* h2    = (half_t*)alloc((size_t)NNET * 2944 * 2);  // [net][32][92]
  half_t* feats = (half_t*)alloc((size_t)NNET * 32 * 2);    // [b*135][32]
  half_t* s_h   = (half_t*)alloc((size_t)BATCH * 9056 * 2); // [64][9056] (pad)
  half_t* w2h   = (half_t*)alloc((size_t)64 * 64 * 2);      // conv2_w K48->64
  half_t* w3h   = (half_t*)alloc((size_t)96 * 96 * 2);      // conv3_w K96
  half_t* fewh  = (half_t*)alloc((size_t)32 * 2176 * 2);    // fe_w  K2160->2176
  half_t* s1wh  = (half_t*)alloc((size_t)32 * 64 * 2);      // s1_w  K64
  half_t* C1wh  = (half_t*)alloc((size_t)1024 * 9056 * 2);  // c1_w  K9045->9056
  half_t* C2wh  = (half_t*)alloc((size_t)256 * 1024 * 2);
  half_t* C3wh  = (half_t*)alloc((size_t)64 * 256 * 2);
  half_t* c1o   = (half_t*)alloc((size_t)64 * 1024 * 2);
  half_t* c2o   = (half_t*)alloc((size_t)64 * 256 * 2);
  half_t* c3o   = (half_t*)alloc((size_t)64 * 64 * 2);
  int*    pii   = (int*)alloc((size_t)NPAIRS * 4);
  int*    pjj   = (int*)alloc((size_t)NPAIRS * 4);
  half_t* h3    = h1;  // alias: h1 is dead once conv2 stage has run; h3 (37.6MB) < h1 (51.4MB)

  // ---- weight conversions (fp32 -> f16, K-padded) ----
  auto cvt = [&](const float* s, half_t* d, int R, int C, int Cp) {
    cvt_pad_kernel<<<(R * Cp + 255) / 256, 256, 0, stream>>>(s, d, R, C, Cp);
  };
  cvt(cw2, w2h, 64, 48, 64);
  cvt(cw3, w3h, 96, 96, 96);
  cvt(few, fewh, 32, 2160, 2176);
  cvt(s1w, s1wh, 32, 64, 64);
  cvt(C1w, C1wh, 1024, 9045, 9056);
  cvt(C2w, C2wh, 256, 1024, 1024);
  cvt(C3w, C3wh, 64, 256, 256);
  pair_idx_kernel<<<(NPAIRS + 255) / 256, 256, 0, stream>>>(pii, pjj);

  // ---- stage 1: conv1 + bn + leaky + pool (VALU) ----
  {
    int total = NNET * 16 * 186;
    conv1_kernel<<<(total + 255) / 256, 256, 0, stream>>>(x, cw1, cb1, g1, bb1, h1);
  }
  // ---- stage 2: conv2 (16->64) as WMMA GEMM, fused bn/leaky/pool ----
  conv_bn_pool_kernel<16, 186, 64, 12, 2, 92>
      <<<NNET, 128, 0, stream>>>(h1, w2h, cb2, g2, bb2, h2, 2944);
  // ---- stage 3: conv3 (32->96) as WMMA GEMM -> h3 (aliases h1) ----
  conv_bn_pool_kernel<32, 92, 96, 6, 3, 45>
      <<<NNET, 192, 0, stream>>>(h2, w3h, cb3, g3, bb3, h3, 2176);
  h3_pad_kernel<<<(NNET * 16 + 255) / 256, 256, 0, stream>>>(h3);
  // ---- stage 4: fe linear 2160->32 (WMMA) ----
  fe_kernel<<<NNET / 16, 64, 0, stream>>>(h3, fewh, feb, feats);
  // ---- stage 5: pair MLP (s1 WMMA, s2-s4 scalar, tanh) ----
  {
    dim3 grid((NPAIRS + 15) / 16, BATCH);
    pair_mlp_kernel<<<grid, 64, 0, stream>>>(feats, pii, pjj, s1wh, s1b,
                                             s2w, s2b, s3w, s3b, s4w, s4b, s_h);
  }
  s_pad_kernel<<<(BATCH * 11 + 63) / 64, 64, 0, stream>>>(s_h);
  // ---- stage 6: classification head GEMMs (WMMA) ----
  mlp_gemm_kernel<<<dim3(16, 4), 128, 0, stream>>>(s_h, C1wh, C1b, c1o,
                                                   9056, 283, 1024, RR_SLOPE);
  mlp_gemm_kernel<<<dim3(4, 4), 128, 0, stream>>>(c1o, C2wh, C2b, c2o,
                                                  1024, 32, 256, 0.0f);
  mlp_gemm_kernel<<<dim3(1, 4), 128, 0, stream>>>(c2o, C3wh, C3b, c3o,
                                                  256, 8, 64, 0.0f);
  // ---- stage 7: c4 + log_softmax ----
  head_kernel<<<1, 64, 0, stream>>>(c3o, C4w, C4b, out);
}
